// TrackAxialAttention_61692910239784
// MI455X (gfx1250) — compile-verified, matrix operand order checked by reading, not run on hardware
//
#include <hip/hip_runtime.h>
#include <hip/hip_bf16.h>

// CDNA5 / gfx1250, wave32. WMMA bf16 16x16x32 GEMM pipeline for
// track-axial attention (LN -> QKV GEMM -> T=4 softmax attention -> out GEMM + residual).
// GEMM tiles are staged with GLOBAL_LOAD_ASYNC_TO_LDS_B128 (ASYNCcnt) so the
// copy of tile k+1 overlaps the WMMA compute on tile k with no VGPR staging.

typedef __attribute__((ext_vector_type(16))) __bf16 v16bf;
typedef __attribute__((ext_vector_type(8)))  float  v8f;

union FragBf16 { v16bf v; uint4 u[2]; };

__device__ __forceinline__ unsigned short f2bf(float f) {
  unsigned int u = __float_as_uint(f);
  u += 0x7FFFu + ((u >> 16) & 1u);          // round-to-nearest-even
  return (unsigned short)(u >> 16);
}
__device__ __forceinline__ unsigned int pack2(float a, float b) {
  return (unsigned int)f2bf(a) | ((unsigned int)f2bf(b) << 16);
}
__device__ __forceinline__ float bflo(unsigned int w) { return __uint_as_float(w << 16); }
__device__ __forceinline__ float bfhi(unsigned int w) { return __uint_as_float(w & 0xffff0000u); }
__device__ __forceinline__ void unpack8(uint4 u, float* f) {
  f[0] = bflo(u.x); f[1] = bfhi(u.x);
  f[2] = bflo(u.y); f[3] = bfhi(u.y);
  f[4] = bflo(u.z); f[5] = bfhi(u.z);
  f[6] = bflo(u.w); f[7] = bfhi(u.w);
}

// Async 16B copy global -> LDS. lds = wave-relative LDS byte address
// (low 32 bits of a generic pointer into __shared__), goff = byte offset
// added to the SGPR base (GVS mode). Tracked by ASYNCcnt.
__device__ __forceinline__ void async_copy16(unsigned lds, const void* base, unsigned goff) {
  asm volatile("global_load_async_to_lds_b128 %0, %1, %2"
               :: "v"(lds), "v"(goff), "s"(base)
               : "memory");
}
__device__ __forceinline__ void wait_async0() {
  asm volatile("s_wait_asynccnt 0x0" ::: "memory");
}

// ---------------------------------------------------------------- weights f32 -> bf16
__global__ __launch_bounds__(256) void f32_to_bf16_kernel(const float* __restrict__ in,
                                                          unsigned short* __restrict__ out,
                                                          int count) {
  int i = blockIdx.x * 256 + threadIdx.x;
  if (i < count) out[i] = f2bf(in[i]);
}

// ---------------------------------------------------------------- LayerNorm(x) -> bf16
// One wave per row of 512; lane owns 16 contiguous elements.
__global__ __launch_bounds__(256) void ln_bf16_kernel(const float* __restrict__ x,
                                                      const float* __restrict__ g,
                                                      const float* __restrict__ bta,
                                                      unsigned short* __restrict__ hout) {
  const int  lane = threadIdx.x & 31;
  const long row  = (long)blockIdx.x * 8 + (threadIdx.x >> 5);
  const float* px = x + row * 512 + lane * 16;

  float v[16];
  const float4* p4 = (const float4*)px;
#pragma unroll
  for (int i = 0; i < 4; ++i) {
    float4 t = p4[i];
    v[4 * i + 0] = t.x; v[4 * i + 1] = t.y; v[4 * i + 2] = t.z; v[4 * i + 3] = t.w;
  }
  float s = 0.f, s2 = 0.f;
#pragma unroll
  for (int i = 0; i < 16; ++i) { s += v[i]; s2 += v[i] * v[i]; }
#pragma unroll
  for (int off = 16; off >= 1; off >>= 1) {
    s  += __shfl_xor(s, off, 32);
    s2 += __shfl_xor(s2, off, 32);
  }
  const float mean = s * (1.0f / 512.0f);
  const float var  = s2 * (1.0f / 512.0f) - mean * mean;
  const float inv  = rsqrtf(var + 1e-5f);

  float gv[16], bv[16];
  const float4* pg = (const float4*)(g   + lane * 16);
  const float4* pb = (const float4*)(bta + lane * 16);
#pragma unroll
  for (int i = 0; i < 4; ++i) {
    float4 tg = pg[i], tb = pb[i];
    gv[4 * i + 0] = tg.x; gv[4 * i + 1] = tg.y; gv[4 * i + 2] = tg.z; gv[4 * i + 3] = tg.w;
    bv[4 * i + 0] = tb.x; bv[4 * i + 1] = tb.y; bv[4 * i + 2] = tb.z; bv[4 * i + 3] = tb.w;
  }
  unsigned int d[8];
#pragma unroll
  for (int i = 0; i < 8; ++i)
    d[i] = pack2((v[2 * i]     - mean) * inv * gv[2 * i]     + bv[2 * i],
                 (v[2 * i + 1] - mean) * inv * gv[2 * i + 1] + bv[2 * i + 1]);
  unsigned short* po = hout + row * 512 + lane * 16;
  *(uint4*)(po)     = make_uint4(d[0], d[1], d[2], d[3]);
  *(uint4*)(po + 8) = make_uint4(d[4], d[5], d[6], d[7]);
}

// ---------------------------------------------------------------- WMMA bf16 GEMM
// C[M,N] = A[M,K] * B[N,K]^T ; A,B bf16 row-major over K.
// Block: 256 threads (8 waves), 128x128 tile, double-buffered LDS, K-step 32.
// Wave grid 4(M) x 2(N); each wave: 2x4 tiles of 16x16 (v_wmma_f32_16x16x32_bf16).
// Tiles staged with async global->LDS copies (ASYNCcnt).
template <bool RESIDUAL>
__global__ __launch_bounds__(256) void gemm_bf16_wmma(const unsigned short* __restrict__ A,
                                                      const unsigned short* __restrict__ B,
                                                      const float* __restrict__ X,
                                                      void* __restrict__ OutP,
                                                      int N, int K) {
  constexpr int PITCH = 80;               // bytes per 32-bf16 LDS row (64B + 16B pad)
  constexpr int BUFSZ = 2 * 128 * PITCH;  // A tile + B tile per stage
  __shared__ unsigned char smem[2 * BUFSZ];

  const int tid  = threadIdx.x;
  const int lane = tid & 31;
  const int w    = tid >> 5;
  const int wm   = w >> 1;                // 0..3 -> M offset wm*32
  const int wn   = w & 1;                 // 0..1 -> N offset wn*64
  const int r    = lane & 15;
  const int half = lane >> 4;

  const long m0 = (long)blockIdx.y * 128;
  const int  n0 = blockIdx.x * 128;
  const int  KT = K >> 5;

  // 512 16B chunks per 128x32 tile (128 rows x 4), 2 per thread
  const int row0 = tid >> 2,         kc0 = tid & 3;
  const int row1 = (tid + 256) >> 2, kc1 = (tid + 256) & 3;

  const unsigned sbase = (unsigned)(size_t)(&smem[0]);  // wave-relative LDS byte addr

  // LDS destinations (buf 0); buf 1 = + BUFSZ
  const unsigned la0 = sbase + (unsigned)(row0 * PITCH + kc0 * 16);
  const unsigned la1 = sbase + (unsigned)(row1 * PITCH + kc1 * 16);
  const unsigned lb0 = la0 + 128 * PITCH;
  const unsigned lb1 = la1 + 128 * PITCH;

  // Global byte offsets at k = 0 (fit easily in i32)
  const unsigned ga0 = (unsigned)(((m0 + row0) * (long)K + kc0 * 8) * 2);
  const unsigned ga1 = (unsigned)(((m0 + row1) * (long)K + kc1 * 8) * 2);
  const unsigned gb0 = (unsigned)((((long)n0 + row0) * K + kc0 * 8) * 2);
  const unsigned gb1 = (unsigned)((((long)n0 + row1) * K + kc1 * 8) * 2);

  // prologue: stage tile 0 into buffer 0
  async_copy16(la0, A, ga0);
  async_copy16(la1, A, ga1);
  async_copy16(lb0, B, gb0);
  async_copy16(lb1, B, gb1);
  wait_async0();
  __syncthreads();

  v8f acc[2][4] = {};

  for (int kt = 0; kt < KT; ++kt) {
    const int  cur  = kt & 1;
    const bool more = (kt + 1) < KT;
    if (more) {
      const unsigned kb   = (unsigned)((kt + 1) << 6);  // 32 bf16 = 64 bytes
      const unsigned bufo = (unsigned)((cur ^ 1) * BUFSZ);
      async_copy16(la0 + bufo, A, ga0 + kb);
      async_copy16(la1 + bufo, A, ga1 + kb);
      async_copy16(lb0 + bufo, B, gb0 + kb);
      async_copy16(lb1 + bufo, B, gb1 + kb);
    }
    const unsigned char* sA = smem + cur * BUFSZ;
    const unsigned char* sB = sA + 128 * PITCH;

    FragBf16 af[2], bfr[4];
#pragma unroll
    for (int mt = 0; mt < 2; ++mt) {
      const int mr = (wm * 32 + mt * 16 + r) * PITCH + half * 16;
      af[mt].u[0] = *(const uint4*)(sA + mr);        // K 0..7 / 8..15
      af[mt].u[1] = *(const uint4*)(sA + mr + 32);   // K 16..23 / 24..31
    }
#pragma unroll
    for (int nt = 0; nt < 4; ++nt) {
      const int nr = (wn * 64 + nt * 16 + r) * PITCH + half * 16;
      bfr[nt].u[0] = *(const uint4*)(sB + nr);
      bfr[nt].u[1] = *(const uint4*)(sB + nr + 32);
    }
#pragma unroll
    for (int mt = 0; mt < 2; ++mt)
#pragma unroll
      for (int nt = 0; nt < 4; ++nt)
        acc[mt][nt] = __builtin_amdgcn_wmma_f32_16x16x32_bf16(
            false, af[mt].v, false, bfr[nt].v, (short)0, acc[mt][nt], false, false);

    if (more) {
      wait_async0();      // this wave's copies into buf^1 have landed in LDS
      __syncthreads();    // everyone's copies landed; everyone done reading buf cur
    }
  }

  // Epilogue. C/D layout: lane = {half -> M+8*half, r -> N}, VGPR i -> M row.
#pragma unroll
  for (int mt = 0; mt < 2; ++mt)
#pragma unroll
    for (int nt = 0; nt < 4; ++nt) {
      const long mb = m0 + wm * 32 + mt * 16 + half * 8;
      const int  nb = n0 + wn * 64 + nt * 16 + r;
      if (RESIDUAL) {
        float* out = (float*)OutP;
#pragma unroll
        for (int i = 0; i < 8; ++i) {
          const long idx = (mb + i) * N + nb;
          out[idx] = X[idx] + acc[mt][nt][i];
        }
      } else {
        unsigned short* out = (unsigned short*)OutP;
#pragma unroll
        for (int i = 0; i < 8; ++i)
          out[(mb + i) * (long)N + nb] = f2bf(acc[mt][nt][i]);
      }
    }
}

// ---------------------------------------------------------------- T=4 attention
// One wave per (b,n). Lane = (head h = lane>>2, d-quarter = lane&3, 16 d's each).
// qkv rows: m = (b*4+t)*8192+n, 1536 cols (q:0..511, k:512..1023, v:1024..1535).
// Loads of 32 lanes at fixed t cover a contiguous 1024B segment (coalesced).
__global__ __launch_bounds__(256) void attn_t4_kernel(const unsigned short* __restrict__ qkv,
                                                      unsigned short* __restrict__ outb) {
  const int lane = threadIdx.x & 31;
  const int gid  = blockIdx.x * 8 + (threadIdx.x >> 5);  // b*8192 + n
  const int b    = gid >> 13;
  const int n    = gid & 8191;
  const int col  = (lane >> 2) * 64 + (lane & 3) * 16;

  long rowq[4];
#pragma unroll
  for (int t = 0; t < 4; ++t)
    rowq[t] = ((long)((b * 4 + t) * 8192 + n)) * 1536;

  float qf[4][16];
#pragma unroll
  for (int t = 0; t < 4; ++t) {
    const uint4* p = (const uint4*)(qkv + rowq[t] + col);
    unpack8(p[0], &qf[t][0]);
    unpack8(p[1], &qf[t][8]);
  }

  float sc[4][4];
#pragma unroll
  for (int t = 0; t < 4; ++t)
#pragma unroll
    for (int s = 0; s < 4; ++s) sc[t][s] = 0.f;

#pragma unroll
  for (int s = 0; s < 4; ++s) {
    float kf[16];
    const uint4* p = (const uint4*)(qkv + rowq[s] + 512 + col);
    unpack8(p[0], &kf[0]);
    unpack8(p[1], &kf[8]);
#pragma unroll
    for (int t = 0; t < 4; ++t) {
      float a = 0.f;
#pragma unroll
      for (int d = 0; d < 16; ++d) a += qf[t][d] * kf[d];
      sc[t][s] += a;
    }
  }
  // reduce partial dot products across the 4 lanes of this head
#pragma unroll
  for (int t = 0; t < 4; ++t)
#pragma unroll
    for (int s = 0; s < 4; ++s) {
      float v = sc[t][s];
      v += __shfl_xor(v, 1, 32);
      v += __shfl_xor(v, 2, 32);
      sc[t][s] = v;
    }

  float aw[4][4];
#pragma unroll
  for (int t = 0; t < 4; ++t) {
    const float s0 = sc[t][0] * 0.125f, s1 = sc[t][1] * 0.125f;
    const float s2 = sc[t][2] * 0.125f, s3 = sc[t][3] * 0.125f;
    const float mx = fmaxf(fmaxf(s0, s1), fmaxf(s2, s3));
    const float e0 = __expf(s0 - mx), e1 = __expf(s1 - mx);
    const float e2 = __expf(s2 - mx), e3 = __expf(s3 - mx);
    const float inv = 1.f / (e0 + e1 + e2 + e3);
    aw[t][0] = e0 * inv; aw[t][1] = e1 * inv; aw[t][2] = e2 * inv; aw[t][3] = e3 * inv;
  }

  float of[4][16];
#pragma unroll
  for (int t = 0; t < 4; ++t)
#pragma unroll
    for (int d = 0; d < 16; ++d) of[t][d] = 0.f;

#pragma unroll
  for (int s = 0; s < 4; ++s) {
    float vf[16];
    const uint4* p = (const uint4*)(qkv + rowq[s] + 1024 + col);
    unpack8(p[0], &vf[0]);
    unpack8(p[1], &vf[8]);
#pragma unroll
    for (int t = 0; t < 4; ++t) {
      const float wgt = aw[t][s];
#pragma unroll
      for (int d = 0; d < 16; ++d) of[t][d] += wgt * vf[d];
    }
  }

#pragma unroll
  for (int t = 0; t < 4; ++t) {
    const long o = ((long)((b * 4 + t) * 8192 + n)) * 512 + col;
    *(uint4*)(outb + o)     = make_uint4(pack2(of[t][0],  of[t][1]),  pack2(of[t][2],  of[t][3]),
                                         pack2(of[t][4],  of[t][5]),  pack2(of[t][6],  of[t][7]));
    *(uint4*)(outb + o + 8) = make_uint4(pack2(of[t][8],  of[t][9]),  pack2(of[t][10], of[t][11]),
                                         pack2(of[t][12], of[t][13]), pack2(of[t][14], of[t][15]));
  }
}

// ---------------------------------------------------------------- launch
extern "C" void kernel_launch(void* const* d_in, const int* in_sizes, int n_in,
                              void* d_out, int out_size, void* d_ws, size_t ws_size,
                              hipStream_t stream) {
  const float* x     = (const float*)d_in[0];   // (4,4,8192,512)
  const float* w_ln  = (const float*)d_in[1];   // (512)
  const float* b_ln  = (const float*)d_in[2];   // (512)
  const float* w_qkv = (const float*)d_in[3];   // (1536,512)
  const float* w_out = (const float*)d_in[4];   // (512,512)
  float* out = (float*)d_out;

  // workspace layout (bytes)
  unsigned char* ws = (unsigned char*)d_ws;
  unsigned short* wqkv_bf = (unsigned short*)(ws);                         // 1536*512*2   = 1.5 MB
  unsigned short* wout_bf = (unsigned short*)(ws + 1572864);               // 512*512*2    = 0.5 MB
  unsigned short* h_bf    = (unsigned short*)(ws + 2097152);               // 131072*512*2 = 128 MB
  unsigned short* qkv_bf  = (unsigned short*)(ws + 2097152 + 134217728);   // 131072*1536*2= 384 MB
  unsigned short* attn_bf = h_bf;  // h is dead after QKV GEMM; alias the region

  constexpr int M = 131072;  // B*T*N

  f32_to_bf16_kernel<<<(1536 * 512) / 256, 256, 0, stream>>>(w_qkv, wqkv_bf, 1536 * 512);
  f32_to_bf16_kernel<<<(512 * 512) / 256, 256, 0, stream>>>(w_out, wout_bf, 512 * 512);

  ln_bf16_kernel<<<M / 8, 256, 0, stream>>>(x, w_ln, b_ln, h_bf);

  gemm_bf16_wmma<false><<<dim3(1536 / 128, M / 128), 256, 0, stream>>>(
      h_bf, wqkv_bf, nullptr, (void*)qkv_bf, 1536, 512);

  attn_t4_kernel<<<(4 * 8192) / 8, 256, 0, stream>>>(qkv_bf, attn_bf);

  gemm_bf16_wmma<true><<<dim3(512 / 128, M / 128), 256, 0, stream>>>(
      attn_bf, wout_bf, x, (void*)out, 512, 512);
}
